// PrePathwayResidualEncoder_7284264534658
// MI455X (gfx1250) — matrix operand order repaired and checked by reading, not compiled, not run on hardware
//
#include <hip/hip_runtime.h>
#include <hip/hip_bf16.h>

typedef _Float16 f16;
typedef __attribute__((ext_vector_type(16))) _Float16 v16h;
typedef __attribute__((ext_vector_type(8)))  _Float16 v8h;
typedef __attribute__((ext_vector_type(8)))  float    v8f;

#define C_DIM 256
#define T_DIM 1024
#define B_DIM 64
#define H_DIM 512
#define TT    64                       // time tile
#define WSTR  (C_DIM + 8)              // half-stride for Wh / St rows (pad kills bank conflicts)
#define MXSTR (128 + 8)                // float-stride for Mx rows

// LDS partition (bytes), all 16B aligned
#define WH_BYTES (128 * WSTR * 2)      // 67584 : W half-tile, f16 [128][WSTR]
#define ST_BYTES (TT  * WSTR * 2)      // 33792 : spike tile, f16 [TT][WSTR] (transposed [t][c])
#define MX_BYTES (TT  * MXSTR * 4)     // 34816 : mixed tile, f32 [TT][MXSTR]
#define SMEM_BYTES (WH_BYTES + ST_BYTES + MX_BYTES)   // 136192 (<160KB -> 2 blocks/WGP)

// d_out layout (flat float32, in reference return order)
#define OUT_D_OFF   0
#define OUT_A_OFF   (B_DIM * H_DIM)                    // 32768
#define OUT_E_OFF   (2 * B_DIM * H_DIM)                // 65536
#define OUT_RX_OFF  (3 * B_DIM * H_DIM)                // 98304
#define OUT_SP_OFF  (OUT_RX_OFF + B_DIM * 2 * C_DIM * T_DIM)  // 33652736

// ---------------------------------------------------------------------------
// Kernel 1: fused  W @ spikes  (WMMA f16->f32)  +  sequential LIF scan.
// grid.x = 192 sequences * 2 channel-halves ; block = 128 threads (4 waves)
// ---------------------------------------------------------------------------
__global__ __launch_bounds__(128) void mix_lif_kernel(
    const float* __restrict__ tx_spikes,   // [B][C][T]
    const float* __restrict__ rx_spikes,   // [B*2][C][T]
    const float* __restrict__ W_tx, const float* __restrict__ b_tx,
    const float* __restrict__ W_rx, const float* __restrict__ b_rx,
    float* __restrict__ out_rx_trace,      // [B*2][C][T]
    float* __restrict__ out_spike_proxy,   // [B*2][C]
    float* __restrict__ ws_fdist)          // [B][C]  (tx time-sums)
{
  extern __shared__ char smem[];
  f16*   Wh = (f16*)(smem);
  f16*   St = (f16*)(smem + WH_BYTES);
  float* Mx = (float*)(smem + WH_BYTES + ST_BYTES);

  const int tid  = threadIdx.x;
  const int wave = tid >> 5;
  const int lane = tid & 31;
  const int l15  = lane & 15;
  const int hiL  = lane >> 4;            // 0 or 1 (upper half of wave)

  const int blk   = blockIdx.x;
  const int s     = blk >> 1;            // sequence 0..191
  const int oBase = (blk & 1) * 128;     // which half of the 256 output channels

  const bool isRx = (s >= B_DIM);
  const float* spikes;
  const float* W;
  const float* bias;
  float* trace = nullptr;
  if (!isRx) {
    spikes = tx_spikes + (size_t)s * C_DIM * T_DIM;
    W = W_tx; bias = b_tx;
  } else {
    const int r = s - B_DIM;             // == b*2 + ear
    spikes = rx_spikes + (size_t)r * C_DIM * T_DIM;
    W = W_rx; bias = b_rx;
    trace = out_rx_trace + (size_t)r * C_DIM * T_DIM;
  }

  // ---- stage W rows [oBase, oBase+128) into LDS as f16 (A operand) ----
  for (int i = tid; i < 128 * C_DIM; i += 128) {
    const int m = i >> 8;
    const int c = i & 255;
    Wh[m * WSTR + c] = (f16)W[(size_t)(oBase + m) * C_DIM + c];
  }
  const float bias_o = bias[oBase + tid];

  // LIF state carried in registers across all time tiles
  float mem = 0.f, spk = 0.f, ssum = 0.f;

  __syncthreads();

  for (int tt = 0; tt < T_DIM / TT; ++tt) {
    const int t0 = tt * TT;

    // ---- stage spike tile transposed: St[t][c] = (f16)spikes[c][t0+t] ----
    for (int i = tid; i < C_DIM * (TT / 4); i += 128) {
      const int c  = i >> 4;
      const int t4 = (i & 15) * 4;
      const float4 v = *(const float4*)(spikes + (size_t)c * T_DIM + t0 + t4);
      St[(t4 + 0) * WSTR + c] = (f16)v.x;
      St[(t4 + 1) * WSTR + c] = (f16)v.y;
      St[(t4 + 2) * WSTR + c] = (f16)v.z;
      St[(t4 + 3) * WSTR + c] = (f16)v.w;
      if (tt + 1 < T_DIM / TT)   // global_prefetch_b8 next tile
        __builtin_prefetch(spikes + (size_t)c * T_DIM + t0 + TT + t4, 0, 0);
    }
    __syncthreads();

    // ---- GEMM tile: Mx[t][m] = sum_c Wh[m][c] * St[t][c] ----
    // wave w owns M rows [w*32, w*32+32) x all 4 N(t)-tiles : 8 accumulators
    const v8f vzero = {0.f, 0.f, 0.f, 0.f, 0.f, 0.f, 0.f, 0.f};
    v8f acc[2][4];
#pragma unroll
    for (int mt = 0; mt < 2; ++mt)
#pragma unroll
      for (int nt = 0; nt < 4; ++nt) acc[mt][nt] = vzero;

#pragma unroll
    for (int kk = 0; kk < C_DIM / 32; ++kk) {
      // A layout (16-bit 16x32): lane<16 -> M=l15, K = kk*32 + {0..7, 16..23}
      //                          lane>=16 -> M=l15, K = kk*32 + {8..15, 24..31}
      const int kha = kk * 32 + hiL * 8;
      v16h afrag[2], bfrag[4];
#pragma unroll
      for (int mt = 0; mt < 2; ++mt) {
        const f16* ap = Wh + (wave * 32 + mt * 16 + l15) * WSTR + kha;
        const v8h lo = *(const v8h*)(ap);
        const v8h hi = *(const v8h*)(ap + 16);
        afrag[mt] = __builtin_shufflevector(lo, hi, 0,1,2,3,4,5,6,7,8,9,10,11,12,13,14,15);
      }
      // B layout (16-bit 32x16): lane<16 -> N=l15, K = kk*32 + {0..7, 16..23}
      //                          lane>=16 -> N=l15, K = kk*32 + {8..15, 24..31}
      const int khb = kk * 32 + hiL * 8;
#pragma unroll
      for (int nt = 0; nt < 4; ++nt) {
        const f16* bp = St + (nt * 16 + l15) * WSTR + khb;
        const v8h lo = *(const v8h*)(bp);
        const v8h hi = *(const v8h*)(bp + 16);
        bfrag[nt] = __builtin_shufflevector(lo, hi, 0,1,2,3,4,5,6,7,8,9,10,11,12,13,14,15);
      }
#pragma unroll
      for (int mt = 0; mt < 2; ++mt)
#pragma unroll
        for (int nt = 0; nt < 4; ++nt)
          acc[mt][nt] = __builtin_amdgcn_wmma_f32_16x16x32_f16(
              false, afrag[mt], false, bfrag[nt], (short)0, acc[mt][nt],
              false, false);
    }

    // ---- spill accumulators to Mx[t][m]; D layout: lane = N(t), VGPR r = M ----
#pragma unroll
    for (int mt = 0; mt < 2; ++mt) {
      const int mb = wave * 32 + mt * 16 + hiL * 8;   // 8 contiguous M per lane
#pragma unroll
      for (int nt = 0; nt < 4; ++nt) {
        float* p = Mx + (nt * 16 + l15) * MXSTR + mb;
        const v8f a = acc[mt][nt];
        *(float4*)(p)     = make_float4(a[0], a[1], a[2], a[3]);
        *(float4*)(p + 4) = make_float4(a[4], a[5], a[6], a[7]);
      }
    }
    __syncthreads();

    // ---- LIF scan over this tile: thread tid <-> local channel o ----
    float4 tr;
#pragma unroll 4
    for (int j = 0; j < TT; ++j) {
      const float x = Mx[j * MXSTR + tid] + bias_o;
      mem = 0.9f * mem + x - spk * 0.75f;
      spk = (mem - 0.75f > 0.f) ? 1.f : 0.f;
      ssum += spk;
      ((float*)&tr)[j & 3] = spk;
      if (isRx && ((j & 3) == 3))
        *(float4*)(trace + (size_t)(oBase + tid) * T_DIM + t0 + (j - 3)) = tr;
    }
    __syncthreads();
  }

  // ---- per-channel time sums ----
  if (!isRx)
    ws_fdist[s * C_DIM + oBase + tid] = ssum;
  else
    out_spike_proxy[(s - B_DIM) * C_DIM + oBase + tid] = ssum;
}

// ---------------------------------------------------------------------------
// Kernel 2: layer-norm over C + three H-dim heads + scaled residual ReLU
// grid = 64 batches, block = 256 threads
// ---------------------------------------------------------------------------
__global__ __launch_bounds__(256) void head_kernel(
    const float* __restrict__ base_d, const float* __restrict__ base_a,
    const float* __restrict__ base_e,
    const float* __restrict__ Wd, const float* __restrict__ bd,
    const float* __restrict__ Wa, const float* __restrict__ ba,
    const float* __restrict__ We, const float* __restrict__ be,
    const float* __restrict__ gd, const float* __restrict__ ga,
    const float* __restrict__ ge,
    const float* __restrict__ ws_fdist, const float* __restrict__ spike_proxy,
    float* __restrict__ out_d, float* __restrict__ out_a, float* __restrict__ out_e)
{
  __shared__ float ln[3][C_DIM];
  __shared__ float red0[256];
  __shared__ float red1[256];

  const int b   = blockIdx.x;
  const int tid = threadIdx.x;
  const float inv_t = 1.f / (float)T_DIM;

  float v[3];
  {
    const float sL = spike_proxy[(b * 2 + 0) * C_DIM + tid];
    const float sR = spike_proxy[(b * 2 + 1) * C_DIM + tid];
    v[0] = ws_fdist[b * C_DIM + tid] * inv_t;   // f_dist
    v[1] = (sL - sR) * inv_t;                   // f_azi
    v[2] = (sL + sR) * inv_t;                   // f_ele
  }

#pragma unroll
  for (int hx = 0; hx < 3; ++hx) {
    red0[tid] = v[hx];
    red1[tid] = v[hx] * v[hx];
    __syncthreads();
    for (int off = 128; off > 0; off >>= 1) {
      if (tid < off) {
        red0[tid] += red0[tid + off];
        red1[tid] += red1[tid + off];
      }
      __syncthreads();
    }
    const float mu  = red0[0] * (1.f / C_DIM);
    const float var = red1[0] * (1.f / C_DIM) - mu * mu;   // population var
    ln[hx][tid] = (v[hx] - mu) * rsqrtf(var + 1e-5f);
    __syncthreads();
  }

  const float sc[3] = {0.3f / (1.f + expf(-gd[0])),
                       0.3f / (1.f + expf(-ga[0])),
                       0.3f / (1.f + expf(-ge[0]))};
  const float* Ws[3]  = {Wd, Wa, We};
  const float* bs[3]  = {bd, ba, be};
  const float* bas[3] = {base_d, base_a, base_e};
  float* outs[3]      = {out_d, out_a, out_e};

#pragma unroll
  for (int hx = 0; hx < 3; ++hx) {
    for (int h = tid; h < H_DIM; h += 256) {
      const float* wr = Ws[hx] + (size_t)h * C_DIM;
      float acc = 0.f;
#pragma unroll 8
      for (int c = 0; c < C_DIM; ++c) acc = __builtin_fmaf(ln[hx][c], wr[c], acc);
      const float r = bas[hx][b * H_DIM + h] + sc[hx] * (acc + bs[hx][h]);
      outs[hx][b * H_DIM + h] = fmaxf(r, 0.f);
    }
  }
}

// ---------------------------------------------------------------------------
extern "C" void kernel_launch(void* const* d_in, const int* in_sizes, int n_in,
                              void* d_out, int out_size, void* d_ws, size_t ws_size,
                              hipStream_t stream) {
  (void)in_sizes; (void)n_in; (void)out_size; (void)ws_size;
  const float* transmit = (const float*)d_in[0];
  const float* receive  = (const float*)d_in[1];
  const float* base_d   = (const float*)d_in[2];
  const float* base_a   = (const float*)d_in[3];
  const float* base_e   = (const float*)d_in[4];
  const float* W_tx     = (const float*)d_in[5];
  const float* b_tx     = (const float*)d_in[6];
  const float* W_rx     = (const float*)d_in[7];
  const float* b_rx     = (const float*)d_in[8];
  const float* Wd       = (const float*)d_in[9];
  const float* bd       = (const float*)d_in[10];
  const float* Wa       = (const float*)d_in[11];
  const float* ba       = (const float*)d_in[12];
  const float* We       = (const float*)d_in[13];
  const float* be       = (const float*)d_in[14];
  const float* gd       = (const float*)d_in[15];
  const float* ga       = (const float*)d_in[16];
  const float* ge       = (const float*)d_in[17];

  float* out    = (float*)d_out;
  float* out_d  = out + OUT_D_OFF;
  float* out_a  = out + OUT_A_OFF;
  float* out_e  = out + OUT_E_OFF;
  float* out_rx = out + OUT_RX_OFF;
  float* out_sp = out + OUT_SP_OFF;
  float* ws_fd  = (float*)d_ws;

  mix_lif_kernel<<<dim3(2 * (B_DIM + 2 * B_DIM)), dim3(128), SMEM_BYTES, stream>>>(
      transmit, receive, W_tx, b_tx, W_rx, b_rx, out_rx, out_sp, ws_fd);

  head_kernel<<<dim3(B_DIM), dim3(256), 0, stream>>>(
      base_d, base_a, base_e, Wd, bd, Wa, ba, We, be, gd, ga, ge,
      ws_fd, out_sp, out_d, out_a, out_e);
}
